// PointNetSetAbstraction_11123965297212
// MI455X (gfx1250) — matrix operand order repaired and checked by reading, not compiled, not run on hardware
//
#include <hip/hip_runtime.h>
#include <hip/hip_bf16.h>

typedef __attribute__((ext_vector_type(16))) _Float16 v16h;
typedef __attribute__((ext_vector_type(8)))  _Float16 v8h;
typedef __attribute__((ext_vector_type(8)))  float    v8f;

// Problem constants (match reference)
#define B_   16
#define N_   4096
#define S_   1024
#define K_   32
#define CPTS 64
#define C0P  96            // 3+64=67 padded to 96 (multiple of WMMA K=32)
#define M_TOT (B_ * S_ * K_)   // 524288
#define EPS_ 1e-5f

// ---------------- workspace layout (bytes, all 256-aligned) ----------------
#define OFF_NN    ((size_t)0)               // int[524288]            = 2097152
#define OFF_WH0   ((size_t)2097152)         // f16 64x96              = 12288
#define OFF_WH1   ((size_t)2109440)         // f16 64x64              = 8192
#define OFF_WH2   ((size_t)2117632)         // f16 128x64             = 16384
#define OFF_STAT  ((size_t)2134016)         // float sum128,sq128,scale128,shift128 = 2048
#define OFF_A0    ((size_t)2136064)         // f16 524288x96          = 100663296
#define OFF_A1    ((size_t)102799360)       // f16 524288x128         = 134217728

// ======================= 1) Farthest point sampling ========================
// One workgroup per batch; running min-distance array lives in LDS (16 KB of
// the 320 KB WGP pool). Sequential loop over S samples (inherent dependence),
// block-wide argmax tree each step.
__global__ void fps_kernel(const float* __restrict__ xyz, float* __restrict__ new_xyz) {
    __shared__ float dist[N_];
    __shared__ float rv[256];
    __shared__ int   ri[256];
    __shared__ int   far;
    const int b = blockIdx.x, tid = threadIdx.x;
    const float* xb = xyz + (size_t)b * N_ * 3;
    for (int i = tid; i < N_; i += 256) dist[i] = 1e10f;
    if (tid == 0) far = 0;
    __syncthreads();
    for (int s = 0; s < S_; ++s) {
        const int f = far;
        const float cx = xb[3*f+0], cy = xb[3*f+1], cz = xb[3*f+2];
        if (tid == 0) {
            float* o = new_xyz + ((size_t)b * S_ + s) * 3;
            o[0] = cx; o[1] = cy; o[2] = cz;
        }
        float vmax = -1.0f; int imax = 0;
        for (int i = tid; i < N_; i += 256) {
            const float dx = xb[3*i+0]-cx, dy = xb[3*i+1]-cy, dz = xb[3*i+2]-cz;
            const float d  = dx*dx + dy*dy + dz*dz;
            const float nd = fminf(dist[i], d);
            dist[i] = nd;
            if (nd > vmax) { vmax = nd; imax = i; }   // strict > keeps first index
        }
        rv[tid] = vmax; ri[tid] = imax;
        __syncthreads();
        for (int off = 128; off > 0; off >>= 1) {
            if (tid < off) {
                const float v2 = rv[tid+off]; const int i2 = ri[tid+off];
                if (v2 > rv[tid] || (v2 == rv[tid] && i2 < ri[tid])) { rv[tid] = v2; ri[tid] = i2; }
            }
            __syncthreads();
        }
        if (tid == 0) far = ri[0];
        __syncthreads();
    }
}

// =========================== 2) kNN (top-32 min) ===========================
// One workgroup per (b,s) query; distances in LDS; 32 sequential argmins.
// xyz per batch is 48 KB -> L2 resident across the 1024 blocks per batch.
__global__ void knn_kernel(const float* __restrict__ xyz, const float* __restrict__ new_xyz,
                           int* __restrict__ nn_idx) {
    __shared__ float d2[N_];
    __shared__ float rv[256];
    __shared__ int   ri[256];
    const int row = blockIdx.x;            // b*S + s
    const int b = row / S_, tid = threadIdx.x;
    const float* xb = xyz + (size_t)b * N_ * 3;
    const float* q  = new_xyz + (size_t)row * 3;
    const float qx = q[0], qy = q[1], qz = q[2];
    for (int i = tid; i < N_; i += 256) {
        const float dx = xb[3*i+0]-qx, dy = xb[3*i+1]-qy, dz = xb[3*i+2]-qz;
        d2[i] = dx*dx + dy*dy + dz*dz;
    }
    __syncthreads();
    for (int kk = 0; kk < K_; ++kk) {
        float vmin = 3.4e38f; int imin = N_ - 1;
        for (int i = tid; i < N_; i += 256) {
            const float v = d2[i];
            if (v < vmin) { vmin = v; imin = i; }
        }
        rv[tid] = vmin; ri[tid] = imin;
        __syncthreads();
        for (int off = 128; off > 0; off >>= 1) {
            if (tid < off) {
                const float v2 = rv[tid+off]; const int i2 = ri[tid+off];
                if (v2 < rv[tid] || (v2 == rv[tid] && i2 < ri[tid])) { rv[tid] = v2; ri[tid] = i2; }
            }
            __syncthreads();
        }
        if (tid == 0) { nn_idx[(size_t)row * K_ + kk] = ri[0]; d2[ri[0]] = 3.4e38f; }
        __syncthreads();
    }
}

// ============ 3) Gather + concat features -> f16 A matrix (pad 96) =========
__global__ void feat_kernel(const float* __restrict__ xyz, const float* __restrict__ points,
                            const float* __restrict__ new_xyz, const int* __restrict__ nn_idx,
                            _Float16* __restrict__ A0) {
    const int m = blockIdx.x * 256 + threadIdx.x;      // < M_TOT (exact)
    const int b   = m / (S_ * K_);
    const int rem = m - b * (S_ * K_);
    const int s   = rem / K_;
    const int row = b * S_ + s;
    const int j   = nn_idx[m];
    const float* p  = xyz + ((size_t)b * N_ + j) * 3;
    const float* q  = new_xyz + (size_t)row * 3;
    const float* pt = points + ((size_t)b * N_ + j) * CPTS;
    _Float16* f = A0 + (size_t)m * C0P;
    f[0] = (_Float16)(p[0] - q[0]);
    f[1] = (_Float16)(p[1] - q[1]);
    f[2] = (_Float16)(p[2] - q[2]);
    #pragma unroll 8
    for (int c = 0; c < CPTS; ++c) f[3 + c] = (_Float16)pt[c];
    #pragma unroll
    for (int c = 3 + CPTS; c < C0P; ++c) f[c] = (_Float16)0.f;
}

// ================= weight convert f32 -> f16 with zero pad =================
__global__ void wconv_kernel(const float* __restrict__ w, _Float16* __restrict__ wh,
                             int O, int Cin, int Cpad) {
    const int t = blockIdx.x * 256 + threadIdx.x;
    if (t >= O * Cpad) return;
    const int o = t / Cpad, c = t - o * Cpad;
    wh[t] = (c < Cin) ? (_Float16)w[o * Cin + c] : (_Float16)0.f;
}

__global__ void zstat_kernel(float* __restrict__ p) { p[threadIdx.x] = 0.f; }  // <<<1,256>>>

// ======================= 4) WMMA GEMM + BN statistics ======================
// One wave owns one 16-row M tile and ALL N tiles: the A fragment is built
// once per K-step (two contiguous b128 loads per lane, per the documented
// wave32 16-bit A layout) and reused for NTILES back-to-back WMMAs.
// BN_IN: apply previous layer's BN scale/shift + ReLU to A elements on load
// (fused epilogue of the previous layer; amortized over NTILES WMMAs, and
// XDL WMMA co-executes with this VALU work). Stats (sum, sum^2 of conv+bias)
// are reduced in LDS then one global atomic per channel per block.
template <int KDIM, int NTILES, bool BN_IN>
__global__ void gemm_wmma_kernel(const _Float16* __restrict__ A, int lda,
                                 const _Float16* __restrict__ W, int ldw,
                                 const float* __restrict__ bias,
                                 const float* __restrict__ iscale,
                                 const float* __restrict__ ishift,
                                 _Float16* __restrict__ Y, int ldy,
                                 float* __restrict__ gsum, float* __restrict__ gsq) {
    __shared__ float lsum[128];
    __shared__ float lsq[128];
    const int tid = threadIdx.x;
    if (tid < 128) { lsum[tid] = 0.f; lsq[tid] = 0.f; }
    __syncthreads();

    const int wave = tid >> 5, lane = tid & 31;          // wave32
    const int mtile = blockIdx.x * 4 + wave;
    const int half = lane >> 4, lq = lane & 15;
    const int row = mtile * 16 + lq;                     // A-tile row for this lane
    const _Float16* ap = A + (size_t)row * lda;

    v8f acc[NTILES];
    #pragma unroll
    for (int nt = 0; nt < NTILES; ++nt) acc[nt] = (v8f){};

    #pragma unroll
    for (int kb = 0; kb < KDIM; kb += 32) {
        // A 16x32 f16 fragment: lanes 0-15 -> K {kb..kb+7, kb+16..kb+23}; lanes 16-31 -> +8
        const v8h alo = *(const v8h*)(ap + kb + 8 * half);
        const v8h ahi = *(const v8h*)(ap + kb + 16 + 8 * half);
        v16h a;
        if (BN_IN) {
            #pragma unroll
            for (int e = 0; e < 8; ++e) {
                const int c0 = kb + 8 * half + e;
                const int c1 = kb + 16 + 8 * half + e;
                a[e]     = (_Float16)fmaxf((float)alo[e] * iscale[c0] + ishift[c0], 0.f);
                a[e + 8] = (_Float16)fmaxf((float)ahi[e] * iscale[c1] + ishift[c1], 0.f);
            }
        } else {
            #pragma unroll
            for (int e = 0; e < 8; ++e) { a[e] = alo[e]; a[e + 8] = ahi[e]; }
        }
        // B 32x16 f16 fragments: column n = nt*16+lq, K same split as A from weight row
        #pragma unroll
        for (int nt = 0; nt < NTILES; ++nt) {
            const _Float16* wp = W + (size_t)(nt * 16 + lq) * ldw + kb + 8 * half;
            const v8h blo = *(const v8h*)(wp);
            const v8h bhi = *(const v8h*)(wp + 16);
            v16h bfrag;
            #pragma unroll
            for (int e = 0; e < 8; ++e) { bfrag[e] = blo[e]; bfrag[e + 8] = bhi[e]; }
            acc[nt] = __builtin_amdgcn_wmma_f32_16x16x32_f16(false, a, false, bfrag,
                                                             (short)0, acc[nt], false, false);
        }
    }

    #pragma unroll
    for (int nt = 0; nt < NTILES; ++nt) {
        const int n = nt * 16 + lq;                      // output channel for this lane
        const float bs = bias[n];
        float s1 = 0.f, s2 = 0.f;
        #pragma unroll
        for (int r = 0; r < 8; ++r) {                    // D layout: m = 16*mtile + r + 8*half
            const float v = acc[nt][r] + bs;
            const int m = mtile * 16 + 8 * half + r;
            Y[(size_t)m * ldy + n] = (_Float16)v;
            s1 += v; s2 += v * v;
        }
        atomicAdd(&lsum[n], s1);
        atomicAdd(&lsq[n],  s2);
    }
    __syncthreads();
    if (tid < NTILES * 16) { atomicAdd(&gsum[tid], lsum[tid]); atomicAdd(&gsq[tid], lsq[tid]); }
}

// ====================== 5) BN finalize (scale/shift) =======================
__global__ void bnfin_kernel(const float* __restrict__ gsum, const float* __restrict__ gsq,
                             const float* __restrict__ gamma, const float* __restrict__ beta,
                             float* __restrict__ scale, float* __restrict__ shift,
                             int Nch, float invM) {
    const int t = threadIdx.x;
    if (t < Nch) {
        const float mean = gsum[t] * invM;
        const float var  = gsq[t] * invM - mean * mean;
        const float sc   = gamma[t] * rsqrtf(var + EPS_);
        scale[t] = sc;
        shift[t] = beta[t] - mean * sc;
    }
}

// ====== 6) fused BN2 + ReLU + max over K neighbors -> [B,128,S] layout =====
__global__ void maxpool_kernel(const _Float16* __restrict__ A,
                               const float* __restrict__ scale, const float* __restrict__ shift,
                               float* __restrict__ out) {
    const int t = blockIdx.x * 256 + threadIdx.x;        // < B*128*S (exact)
    const int b  = t >> 17;                              // / (128*1024)
    const int r  = t & 131071;
    const int ch = r >> 10;
    const int s  = r & 1023;
    const float sc = scale[ch], sh = shift[ch];
    const _Float16* base = A + ((size_t)(b * S_ + s) * K_) * 128 + ch;
    float m = 0.0f;                                      // ReLU output is >= 0
    #pragma unroll 8
    for (int k = 0; k < K_; ++k)
        m = fmaxf(m, fmaxf((float)base[k * 128] * sc + sh, 0.f));
    out[t] = m;                                          // out layout == [B,128,S] flat
}

// ================================ launch ===================================
extern "C" void kernel_launch(void* const* d_in, const int* in_sizes, int n_in,
                              void* d_out, int out_size, void* d_ws, size_t ws_size,
                              hipStream_t stream) {
    const float* xyz    = (const float*)d_in[0];
    const float* points = (const float*)d_in[1];
    const float* w0 = (const float*)d_in[2],  *b0 = (const float*)d_in[3];
    const float* g0 = (const float*)d_in[4],  *be0 = (const float*)d_in[5];
    const float* w1 = (const float*)d_in[6],  *b1 = (const float*)d_in[7];
    const float* g1 = (const float*)d_in[8],  *be1 = (const float*)d_in[9];
    const float* w2 = (const float*)d_in[10], *b2 = (const float*)d_in[11];
    const float* g2 = (const float*)d_in[12], *be2 = (const float*)d_in[13];

    char* ws = (char*)d_ws;
    int*      nn    = (int*)(ws + OFF_NN);
    _Float16* Wh0   = (_Float16*)(ws + OFF_WH0);
    _Float16* Wh1   = (_Float16*)(ws + OFF_WH1);
    _Float16* Wh2   = (_Float16*)(ws + OFF_WH2);
    float*    gsum  = (float*)(ws + OFF_STAT);
    float*    gsq   = gsum + 128;
    float*    scale = gsum + 256;
    float*    shift = gsum + 384;
    _Float16* A0    = (_Float16*)(ws + OFF_A0);   // feat (stride 96) / layer1 out (stride 64)
    _Float16* A1    = (_Float16*)(ws + OFF_A1);   // layer0 out (stride 64) / layer2 out (stride 128)

    float* new_xyz = (float*)d_out;               // [B,S,3]
    float* out_pts = (float*)d_out + (size_t)B_ * S_ * 3;  // [B,128,S]

    // Sampling / grouping
    fps_kernel<<<B_, 256, 0, stream>>>(xyz, new_xyz);
    knn_kernel<<<B_ * S_, 256, 0, stream>>>(xyz, new_xyz, nn);
    wconv_kernel<<<(64 * 96 + 255) / 256, 256, 0, stream>>>(w0, Wh0, 64, 67, 96);
    wconv_kernel<<<(64 * 64 + 255) / 256, 256, 0, stream>>>(w1, Wh1, 64, 64, 64);
    wconv_kernel<<<(128 * 64 + 255) / 256, 256, 0, stream>>>(w2, Wh2, 128, 64, 64);
    feat_kernel<<<M_TOT / 256, 256, 0, stream>>>(xyz, points, new_xyz, nn, A0);

    const int MT = M_TOT / 16;                    // 32768 M-tiles; 4 waves per block
    const float invM = 1.0f / (float)M_TOT;

    // Layer 0: 96(padded 67) -> 64  (raw features in; pre-BN f16 out)
    zstat_kernel<<<1, 256, 0, stream>>>(gsum);
    gemm_wmma_kernel<96, 4, false><<<MT / 4, 128, 0, stream>>>(
        A0, 96, Wh0, 96, b0, nullptr, nullptr, A1, 64, gsum, gsq);
    bnfin_kernel<<<1, 128, 0, stream>>>(gsum, gsq, g0, be0, scale, shift, 64, invM);

    // Layer 1: 64 -> 64  (BN0+ReLU fused into A load)
    zstat_kernel<<<1, 256, 0, stream>>>(gsum);
    gemm_wmma_kernel<64, 4, true><<<MT / 4, 128, 0, stream>>>(
        A1, 64, Wh1, 64, b1, scale, shift, A0, 64, gsum, gsq);
    bnfin_kernel<<<1, 128, 0, stream>>>(gsum, gsq, g1, be1, scale, shift, 64, invM);

    // Layer 2: 64 -> 128  (BN1+ReLU fused into A load)
    zstat_kernel<<<1, 256, 0, stream>>>(gsum);
    gemm_wmma_kernel<64, 8, true><<<MT / 4, 128, 0, stream>>>(
        A0, 64, Wh2, 64, b2, scale, shift, A1, 128, gsum, gsq);
    bnfin_kernel<<<1, 128, 0, stream>>>(gsum, gsq, g2, be2, scale, shift, 128, invM);

    // Fused BN2 + ReLU + max over K neighbors -> [B,128,S]
    maxpool_kernel<<<(B_ * 128 * S_) / 256, 256, 0, stream>>>(A1, scale, shift, out_pts);
}
